// DecoderOnlyTransformer_83854941487731
// MI455X (gfx1250) — matrix operand order repaired and checked by reading, not compiled, not run on hardware
//
#include <hip/hip_runtime.h>
#include <hip/hip_bf16.h>
#include <math.h>

// ---------------- model dims (GPT-2 small, B=2, S=1024) ----------------
constexpr int Vv = 50257;
constexpr int D  = 768;
constexpr int H  = 12;
constexpr int NL = 12;
constexpr int HD = 64;     // D/H
constexpr int Bb = 2;
constexpr int S  = 1024;
constexpr int BS = Bb * S; // 2048 rows
constexpr float LN_EPS = 1e-5f;

#define DEV static __device__ __forceinline__

typedef __bf16 v16bf __attribute__((ext_vector_type(16)));
typedef __bf16 v8bf  __attribute__((ext_vector_type(8)));
typedef float  v8f   __attribute__((ext_vector_type(8)));
typedef unsigned v4u __attribute__((ext_vector_type(4)));
typedef int      v8i __attribute__((ext_vector_type(8)));
typedef int      v4i __attribute__((ext_vector_type(4)));

#if __has_builtin(__builtin_amdgcn_tensor_load_to_lds)
#define HAVE_TDM 1
#else
#define HAVE_TDM 0
#endif

// f32 -> bf16 using the hardware conversion path
DEV __bf16 f2bf(float f) { return (__bf16)f; }

DEV v8f wmma_bf16(v16bf a, v16bf b, v8f c) {
  // D(f32,16x16) = A(16x32 bf16) * B(32x16 bf16) + C
  return __builtin_amdgcn_wmma_f32_16x16x32_bf16(
      /*neg_a=*/false, a, /*neg_b=*/false, b,
      /*c_mod=*/(short)0, c, /*reuse_a=*/false, /*reuse_b=*/false);
}

// tanh-approx GELU via exp (single transcendental)
DEV float gelu_tanh(float x) {
  float u = 0.7978845608028654f * (x + 0.044715f * x * x * x);
  float t = 1.0f - 2.0f / (__expf(2.0f * u) + 1.0f);  // tanh(u)
  return 0.5f * x * (1.0f + t);
}

#if HAVE_TDM
// TDM 2D tile load: tile_rows rows of 32 bf16, global row stride = stride_elems,
// LDS gets +16B pad after every 64B (-> 40-bf16 padded rows). Hardware
// zero-fills reads beyond (rem_cols x rem_rows), which handles edge tiles.
DEV void tdm_load_tile_bf16(const __bf16* gsrc, unsigned lds_byte_off,
                            int rem_cols, int rem_rows, int tile_rows,
                            int stride_elems) {
  unsigned long long ga = (unsigned long long)(size_t)gsrc;
  v4u g0;
  g0[0] = 1u;                                      // count=1 (valid user D#)
  g0[1] = lds_byte_off;                            // lds_addr
  g0[2] = (unsigned)(ga & 0xFFFFFFFFull);          // global_addr[31:0]
  g0[3] = (unsigned)((ga >> 32) & 0x01FFFFFFull)   // global_addr[56:32]
        | (2u << 30);                              // type = 2 ("image")
  v8i g1;
  g1[0] = (int)((1u << 16)      // data_size = 1 -> 2 bytes
              | (1u << 20)      // pad_enable
              | (3u << 22)      // pad_interval: 16 DWORDs (64B)
              | (3u << 25));    // pad_amount:   4 DWORDs (16B)
  unsigned td0 = (unsigned)rem_cols;   // tensor_dim0 (elements)
  unsigned td1 = (unsigned)rem_rows;   // tensor_dim1 (rows)
  g1[1] = (int)((td0 & 0xFFFFu) << 16);                  // [63:48] = td0.lo16
  g1[2] = (int)((td0 >> 16) | ((td1 & 0xFFFFu) << 16));  // td0.hi16 | td1.lo16
  g1[3] = (int)((td1 >> 16) | (32u << 16));              // td1.hi16 | tile_dim0=32
  g1[4] = (int)(unsigned)tile_rows;                      // tile_dim1 (tile_dim2=0)
  g1[5] = (int)(unsigned)stride_elems;                   // tensor_dim0_stride lo32
  g1[6] = 0;
  g1[7] = 0;
  v4i z4 = {0, 0, 0, 0};
#if __clang_major__ >= 23
  v8i z8 = {0, 0, 0, 0, 0, 0, 0, 0};
  __builtin_amdgcn_tensor_load_to_lds(g0, g1, z4, z4, z8, 0);
#else
  __builtin_amdgcn_tensor_load_to_lds(g0, g1, z4, z4, 0);
#endif
}

DEV void tdm_wait() {
#if __has_builtin(__builtin_amdgcn_s_wait_tensorcnt)
  __builtin_amdgcn_s_wait_tensorcnt(0);
#else
  asm volatile("s_wait_tensorcnt 0x0" ::: "memory");
#endif
}
#endif // HAVE_TDM

// ---------------- elementwise f32 -> bf16 ----------------
__global__ void k_f32_to_bf16(const float* __restrict__ in, __bf16* __restrict__ out, long n) {
  long i = (long)blockIdx.x * blockDim.x + threadIdx.x;
  long stride = (long)gridDim.x * blockDim.x;
  for (; i < n; i += stride) out[i] = f2bf(in[i]);
}

// ---------------- token + position embedding ----------------
__global__ __launch_bounds__(256) void k_embed(const int* __restrict__ x,
                                               const float* __restrict__ wte,
                                               const float* __restrict__ wpe,
                                               float* __restrict__ h) {
  int t = blockIdx.x;        // 0..BS-1
  int s = t % S;
  int id = x[t];
  for (int d0 = threadIdx.x; d0 < D; d0 += 256)
    h[(size_t)t * D + d0] = wte[(size_t)id * D + d0] + wpe[(size_t)s * D + d0];
}

// ---------------- layernorm (f32 in, bf16 out) ----------------
__global__ __launch_bounds__(256) void k_layernorm(const float* __restrict__ xr,
                                                   const float* __restrict__ g,
                                                   const float* __restrict__ bv,
                                                   __bf16* __restrict__ out) {
  int row = blockIdx.x;
  const float* x = xr + (size_t)row * D;
  float vals[3];
  float s1 = 0.f, s2 = 0.f;
#pragma unroll
  for (int i = 0; i < 3; i++) {
    float v = x[threadIdx.x + 256 * i];
    vals[i] = v; s1 += v; s2 += v * v;
  }
#pragma unroll
  for (int off = 1; off < 32; off <<= 1) {
    s1 += __shfl_xor(s1, off, 32);
    s2 += __shfl_xor(s2, off, 32);
  }
  __shared__ float r1[8], r2[8];
  int w = threadIdx.x >> 5, ln = threadIdx.x & 31;
  if (ln == 0) { r1[w] = s1; r2[w] = s2; }
  __syncthreads();
  float t1 = 0.f, t2 = 0.f;
#pragma unroll
  for (int i = 0; i < 8; i++) { t1 += r1[i]; t2 += r2[i]; }
  float mean = t1 / (float)D;
  float var  = t2 / (float)D - mean * mean;
  float rstd = rsqrtf(var + LN_EPS);
#pragma unroll
  for (int i = 0; i < 3; i++) {
    int c = threadIdx.x + 256 * i;
    out[(size_t)row * D + c] = f2bf((vals[i] - mean) * rstd * g[c] + bv[c]);
  }
}

// ---------------- V transpose: qkv bf16 [BS,3D] -> vt [B,H,HD,S] ----------
__global__ __launch_bounds__(256) void k_vtrans(const __bf16* __restrict__ qkv,
                                                __bf16* __restrict__ vt) {
  int t = blockIdx.x;            // token 0..BS-1
  int b = t / S, s = t % S;
  const __bf16* src = qkv + (size_t)t * (3 * D) + 2 * D;
#pragma unroll
  for (int it = 0; it < 3; it++) {
    int dd = threadIdx.x + 256 * it;
    int h = dd >> 6, d = dd & 63;
    vt[((size_t)(b * H + h) * HD + d) * S + s] = src[dd];
  }
}

// ---------------- tiled bf16 WMMA GEMM ----------------
// C[M,N] = A[M,K] * B + epilogue.  A: row-major bf16 (activations).
// B_NMAJOR=false: B is [K,N] row-major (weights). true: B is [N,K] (wte for lm_head).
enum { EP_BIAS = 1, EP_RESID = 2, EP_GELU = 4, EP_OUTF = 8, EP_OUTB = 16 };

template <int FLAGS, bool B_NMAJOR>
__global__ __launch_bounds__(256) void k_gemm(const __bf16* __restrict__ A,
                                              const __bf16* __restrict__ Bm,
                                              const float* __restrict__ bias,
                                              const float* __restrict__ resid,
                                              float* __restrict__ Cf,
                                              __bf16* __restrict__ Cb,
                                              int M, int N, int K) {
  // padded rows (40 bf16 = 80B) -> conflict-free 16B ds_loads; matches TDM pad
  __shared__ __bf16 sA[64][40];    // [m][k]
  __shared__ __bf16 sB[128][40];   // [n][k]
  const int tid  = threadIdx.x;
  const int lane = tid & 31;
  const int wv   = tid >> 5;
  const int ln   = lane & 15;      // A: row M, B: col N within tile
  const int kh   = lane >> 4;      // half-wave selector for K split
  const int wm   = wv >> 2;        // 0..1  (wave M tile of 32)
  const int wn   = wv & 3;         // 0..3  (wave N tile of 32)
  const int m0 = blockIdx.y * 64;
  const int n0 = blockIdx.x * 128;

  v8f acc[2][2] = {};

  for (int k0 = 0; k0 < K; k0 += 32) {
#if HAVE_TDM
    // A tile (and N-major B tile) via Tensor Data Mover: one wave issues the
    // descriptor, DMA does the strided walk + LDS padding; OOB rows zero-fill.
    if (wv == 0) {
      tdm_load_tile_bf16(A + (size_t)m0 * K + k0,
                         (unsigned)(size_t)&sA[0][0],
                         K - k0, M - m0, 64, K);
      if (B_NMAJOR)
        tdm_load_tile_bf16(Bm + (size_t)n0 * K + k0,
                           (unsigned)(size_t)&sB[0][0],
                           K - k0, N - n0, 128, K);
      tdm_wait();
    }
#else
    { // stage A tile: 64x32, one uint4 per thread
      int r = tid >> 2, kq = (tid & 3) * 8;
      uint4 v = *(const uint4*)(A + (size_t)(m0 + r) * K + k0 + kq);
      *(uint4*)(&sA[r][kq]) = v;
    }
    if (B_NMAJOR) { // wte: [N,K]; contiguous along K -> direct vector stage
      int c = tid >> 1, kq = (tid & 1) * 16;
      uint4 v0{0,0,0,0}, v1{0,0,0,0};
      if (n0 + c < N) {
        const __bf16* src = Bm + (size_t)(n0 + c) * K + k0 + kq;
        v0 = *(const uint4*)(src);
        v1 = *(const uint4*)(src + 8);
      }
      *(uint4*)(&sB[c][kq])     = v0;
      *(uint4*)(&sB[c][kq + 8]) = v1;
    }
#endif
    if (!B_NMAJOR) { // weights: [K,N]; vector read along N, scatter to [n][k]
      int kr = tid >> 3, c0 = (tid & 7) * 16;
      const __bf16* src = Bm + (size_t)(k0 + kr) * N + n0 + c0;
      uint4 v0 = *(const uint4*)(src);
      uint4 v1 = *(const uint4*)(src + 8);
      const __bf16* e0 = (const __bf16*)&v0;
      const __bf16* e1 = (const __bf16*)&v1;
#pragma unroll
      for (int i = 0; i < 8; i++) sB[c0 + i][kr] = e0[i];
#pragma unroll
      for (int i = 0; i < 8; i++) sB[c0 + 8 + i][kr] = e1[i];
    }
    __syncthreads();

    // A-fragment: lane m=ln; K pairs: [kh*8 .. kh*8+7] and [16+kh*8 .. +7]
    v16bf af[2], bfr[2];
#pragma unroll
    for (int t = 0; t < 2; t++) {
      int rm = wm * 32 + t * 16 + ln;
      v8bf lo = *(const v8bf*)(&sA[rm][kh * 8]);
      v8bf hi = *(const v8bf*)(&sA[rm][16 + kh * 8]);
      af[t] = __builtin_shufflevector(lo, hi, 0,1,2,3,4,5,6,7,8,9,10,11,12,13,14,15);
    }
    // B-fragment: lane n=ln; K pairs: [kh*8 .. kh*8+7] and [16+kh*8 .. +7]
#pragma unroll
    for (int t = 0; t < 2; t++) {
      int cn = wn * 32 + t * 16 + ln;
      v8bf lo = *(const v8bf*)(&sB[cn][kh * 8]);
      v8bf hi = *(const v8bf*)(&sB[cn][16 + kh * 8]);
      bfr[t] = __builtin_shufflevector(lo, hi, 0,1,2,3,4,5,6,7,8,9,10,11,12,13,14,15);
    }
#pragma unroll
    for (int tm = 0; tm < 2; tm++)
#pragma unroll
      for (int tn = 0; tn < 2; tn++)
        acc[tm][tn] = wmma_bf16(af[tm], bfr[tn], acc[tm][tn]);
    __syncthreads();
  }

  // epilogue: C layout -> element (m = i + 8*kh, n = ln) in VGPR i
#pragma unroll
  for (int tm = 0; tm < 2; tm++) {
#pragma unroll
    for (int tn = 0; tn < 2; tn++) {
      int n = n0 + wn * 32 + tn * 16 + ln;
      if (n >= N) continue;
      float bn = (FLAGS & EP_BIAS) ? bias[n] : 0.f;
#pragma unroll
      for (int i = 0; i < 8; i++) {
        int m = m0 + wm * 32 + tm * 16 + i + 8 * kh;
        float v = acc[tm][tn][i] + bn;
        if (FLAGS & EP_RESID) v += resid[(size_t)m * N + n];
        if (FLAGS & EP_GELU)  v = gelu_tanh(v);
        if (FLAGS & EP_OUTF)  Cf[(size_t)m * N + n] = v;
        if (FLAGS & EP_OUTB)  Cb[(size_t)m * N + n] = f2bf(v);
      }
    }
  }
}

// ---------------- flash attention (one wave per 16-query tile per (b,h)) ----
// qkv: bf16 [BS, 3D];  vt: bf16 [B,H,HD,S]  (V transposed for contiguous key loads)
__global__ __launch_bounds__(32) void k_flash_attn(const __bf16* __restrict__ qkv,
                                                   const __bf16* __restrict__ vt,
                                                   __bf16* __restrict__ out) {
  __shared__ __bf16 sP[16][32];
  const int lane = threadIdx.x;
  const int ln = lane & 15, hi = lane >> 4;
  const int q0 = blockIdx.x * 16;
  const int bh = blockIdx.y;
  const int b = bh / H, h = bh % H;
  const float scale = 0.125f; // 1/sqrt(HD)

  // Q fragments: A-layout over d; both K-runs are contiguous bf16 -> vector loads
  const __bf16* qrow = qkv + (size_t)(b * S + q0 + ln) * (3 * D) + h * HD;
  v16bf qf[2];
#pragma unroll
  for (int f = 0; f < 2; f++) {
    v8bf lo = *(const v8bf*)(qrow + f * 32 + hi * 8);
    v8bf hh = *(const v8bf*)(qrow + f * 32 + 16 + hi * 8);
    qf[f] = __builtin_shufflevector(lo, hh, 0,1,2,3,4,5,6,7,8,9,10,11,12,13,14,15);
  }

  float m_i[8], l_i[8];
#pragma unroll
  for (int i = 0; i < 8; i++) { m_i[i] = -1e30f; l_i[i] = 0.f; }
  v8f o_acc[4] = {};

  const int kend = q0 + 16;                 // causal: keys <= q0+15
  for (int j0 = 0; j0 < kend; j0 += 32) {
    // S = Q * K^T for 32 keys (two 16-wide n-tiles)
    v8f sacc[2] = {};
#pragma unroll
    for (int nt = 0; nt < 2; nt++) {
      int key = j0 + nt * 16 + ln;
      const __bf16* krow = qkv + (size_t)(b * S + key) * (3 * D) + D + h * HD;
#pragma unroll
      for (int f = 0; f < 2; f++) {
        v8bf lo = *(const v8bf*)(krow + f * 32 + hi * 8);
        v8bf hh = *(const v8bf*)(krow + f * 32 + 16 + hi * 8);
        v16bf kf = __builtin_shufflevector(lo, hh, 0,1,2,3,4,5,6,7,8,9,10,11,12,13,14,15);
        sacc[nt] = wmma_bf16(qf[f], kf, sacc[nt]);
      }
    }
    // online softmax; row stats indexed by acc slot i (row = i + 8*hi)
    float alpha[8];
#pragma unroll
    for (int i = 0; i < 8; i++) {
      int q = q0 + i + 8 * hi;
      float v0 = (j0 + ln      <= q) ? sacc[0][i] * scale : -1e30f;
      float v1 = (j0 + 16 + ln <= q) ? sacc[1][i] * scale : -1e30f;
      float mx = fmaxf(v0, v1);
#pragma unroll
      for (int off = 1; off < 16; off <<= 1) mx = fmaxf(mx, __shfl_xor(mx, off, 32));
      float mnew = fmaxf(m_i[i], mx);
      alpha[i] = __expf(m_i[i] - mnew);
      float p0 = __expf(v0 - mnew);
      float p1 = __expf(v1 - mnew);
      sP[i + 8 * hi][ln]      = f2bf(p0);
      sP[i + 8 * hi][16 + ln] = f2bf(p1);
      float rs = p0 + p1;
#pragma unroll
      for (int off = 1; off < 16; off <<= 1) rs += __shfl_xor(rs, off, 32);
      l_i[i] = l_i[i] * alpha[i] + rs;
      m_i[i] = mnew;
    }
#pragma unroll
    for (int t = 0; t < 4; t++)
#pragma unroll
      for (int i = 0; i < 8; i++) o_acc[t][i] *= alpha[i];

    asm volatile("s_wait_dscnt 0x0" ::: "memory");

    // P back out of LDS in A-fragment layout (K = 32 keys)
    v8bf plo = *(const v8bf*)(&sP[ln][hi * 8]);
    v8bf phi = *(const v8bf*)(&sP[ln][16 + hi * 8]);
    v16bf pf = __builtin_shufflevector(plo, phi, 0,1,2,3,4,5,6,7,8,9,10,11,12,13,14,15);

    // O += P * V ; vt gives contiguous keys per (d) row -> vector loads
#pragma unroll
    for (int t = 0; t < 4; t++) {
      const __bf16* vrow = vt + ((size_t)(b * H + h) * HD + t * 16 + ln) * S + j0 + hi * 8;
      v8bf lo = *(const v8bf*)(vrow);
      v8bf hh = *(const v8bf*)(vrow + 16);
      v16bf vf = __builtin_shufflevector(lo, hh, 0,1,2,3,4,5,6,7,8,9,10,11,12,13,14,15);
      o_acc[t] = wmma_bf16(pf, vf, o_acc[t]);
    }
  }

#pragma unroll
  for (int t = 0; t < 4; t++) {
#pragma unroll
    for (int i = 0; i < 8; i++) {
      int srow = q0 + i + 8 * hi;
      out[(size_t)(b * S + srow) * D + h * HD + t * 16 + ln] = f2bf(o_acc[t][i] / l_i[i]);
    }
  }
}

// ---------------- host orchestration ----------------
extern "C" void kernel_launch(void* const* d_in, const int* in_sizes, int n_in,
                              void* d_out, int out_size, void* d_ws, size_t ws_size,
                              hipStream_t stream) {
  (void)in_sizes; (void)n_in; (void)out_size; (void)ws_size;
  const int*   x    = (const int*)  d_in[0];
  const float* wte  = (const float*)d_in[1];
  const float* wpe  = (const float*)d_in[2];
  const float* ln1g = (const float*)d_in[3];
  const float* ln1b = (const float*)d_in[4];
  const float* qkvw = (const float*)d_in[5];
  const float* qkvb = (const float*)d_in[6];
  const float* apw  = (const float*)d_in[7];
  const float* apb  = (const float*)d_in[8];
  const float* ln2g = (const float*)d_in[9];
  const float* ln2b = (const float*)d_in[10];
  const float* fcw  = (const float*)d_in[11];
  const float* fcb  = (const float*)d_in[12];
  const float* mpw  = (const float*)d_in[13];
  const float* mpb  = (const float*)d_in[14];
  const float* lnfg = (const float*)d_in[15];
  const float* lnfb = (const float*)d_in[16];
  float* logits = (float*)d_out;

  char* p = (char*)d_ws;
  auto carve = [&](size_t bytes) -> void* {
    void* r = (void*)p;
    p += (bytes + 255) & ~(size_t)255;
    return r;
  };
  const size_t nQKVW = (size_t)NL * D * 3 * D;
  const size_t nAPW  = (size_t)NL * D * D;
  const size_t nFCW  = (size_t)NL * D * 4 * D;
  const size_t nMPW  = (size_t)NL * 4 * D * D;
  const size_t nWTE  = (size_t)Vv * D;

  __bf16* b_qkvw = (__bf16*)carve(nQKVW * 2);
  __bf16* b_apw  = (__bf16*)carve(nAPW * 2);
  __bf16* b_fcw  = (__bf16*)carve(nFCW * 2);
  __bf16* b_mpw  = (__bf16*)carve(nMPW * 2);
  __bf16* b_wte  = (__bf16*)carve(nWTE * 2);
  float*  hbuf   = (float*) carve((size_t)BS * D * 4);
  __bf16* lnbuf  = (__bf16*)carve((size_t)BS * D * 2);
  __bf16* qkvbf  = (__bf16*)carve((size_t)BS * 3 * D * 2);
  __bf16* vtbuf  = (__bf16*)carve((size_t)BS * D * 2);
  __bf16* attnb  = (__bf16*)carve((size_t)BS * D * 2);
  __bf16* fcbuf  = (__bf16*)carve((size_t)BS * 4 * D * 2);

  // weights -> bf16 (deterministic; same result every call)
  k_f32_to_bf16<<<512, 256, 0, stream>>>(qkvw, b_qkvw, (long)nQKVW);
  k_f32_to_bf16<<<512, 256, 0, stream>>>(apw,  b_apw,  (long)nAPW);
  k_f32_to_bf16<<<512, 256, 0, stream>>>(fcw,  b_fcw,  (long)nFCW);
  k_f32_to_bf16<<<512, 256, 0, stream>>>(mpw,  b_mpw,  (long)nMPW);
  k_f32_to_bf16<<<512, 256, 0, stream>>>(wte,  b_wte,  (long)nWTE);

  k_embed<<<BS, 256, 0, stream>>>(x, wte, wpe, hbuf);

  dim3 gq((3 * D) / 128, BS / 64);   // qkv:  N=2304
  dim3 gp(D / 128,       BS / 64);   // proj: N=768
  dim3 gf((4 * D) / 128, BS / 64);   // fc:   N=3072
  for (int l = 0; l < NL; l++) {
    k_layernorm<<<BS, 256, 0, stream>>>(hbuf, ln1g + l * D, ln1b + l * D, lnbuf);
    k_gemm<EP_BIAS | EP_OUTB, false><<<gq, 256, 0, stream>>>(
        lnbuf, b_qkvw + (size_t)l * D * 3 * D, qkvb + (size_t)l * 3 * D,
        nullptr, nullptr, qkvbf, BS, 3 * D, D);
    k_vtrans<<<BS, 256, 0, stream>>>(qkvbf, vtbuf);
    k_flash_attn<<<dim3(S / 16, Bb * H), 32, 0, stream>>>(qkvbf, vtbuf, attnb);
    k_gemm<EP_BIAS | EP_RESID | EP_OUTF, false><<<gp, 256, 0, stream>>>(
        attnb, b_apw + (size_t)l * D * D, apb + (size_t)l * D,
        hbuf, hbuf, nullptr, BS, D, D);
    k_layernorm<<<BS, 256, 0, stream>>>(hbuf, ln2g + l * D, ln2b + l * D, lnbuf);
    k_gemm<EP_BIAS | EP_GELU | EP_OUTB, false><<<gf, 256, 0, stream>>>(
        lnbuf, b_fcw + (size_t)l * D * 4 * D, fcb + (size_t)l * 4 * D,
        nullptr, nullptr, fcbuf, BS, 4 * D, D);
    k_gemm<EP_BIAS | EP_RESID | EP_OUTF, false><<<gp, 256, 0, stream>>>(
        fcbuf, b_mpw + (size_t)l * 4 * D * D, mpb + (size_t)l * D,
        hbuf, hbuf, nullptr, BS, D, 4 * D);
  }
  k_layernorm<<<BS, 256, 0, stream>>>(hbuf, lnfg, lnfb, lnbuf);

  // tied lm_head: logits = lnf(h) @ wte^T, B is N-major (wte is [V, D])
  dim3 gl((Vv + 127) / 128, BS / 64);
  k_gemm<EP_OUTF, true><<<gl, 256, 0, stream>>>(
      lnbuf, b_wte, nullptr, nullptr, logits, nullptr, BS, Vv, D);
}
